// LLaDA2MoeAttention_27161373179910
// MI455X (gfx1250) — compile-verified, matrix operand order checked by reading, not run on hardware
//
#include <hip/hip_runtime.h>
#include <hip/hip_bf16.h>
#include <math.h>

// ---------------------------------------------------------------------------
// Types
// ---------------------------------------------------------------------------
typedef __bf16 bf16;
typedef __attribute__((ext_vector_type(4)))  bf16  v4bf;
typedef __attribute__((ext_vector_type(8)))  bf16  v8bf;
typedef __attribute__((ext_vector_type(16))) bf16  v16bf;
typedef __attribute__((ext_vector_type(8)))  float v8f;

// ---------------------------------------------------------------------------
// Problem constants
// ---------------------------------------------------------------------------
#define B_   2
#define S_   2048
#define HID_ 2048
#define H_   16
#define HK_  4
#define D_   128
#define R_   64
#define NQKV (H_ + 2*HK_)        // 24 heads in qkv
#define OQKV (NQKV * D_)         // 3072
#define MROWS (B_ * S_)          // 4096
#define INV_SQRT_D 0.08838834764831845f

// ---------------------------------------------------------------------------
// Helpers
// ---------------------------------------------------------------------------
__device__ __forceinline__ v8f zero8() {
    v8f z;
#pragma unroll
    for (int i = 0; i < 8; ++i) z[i] = 0.0f;
    return z;
}

// Load a 16x32 bf16 A/B WMMA fragment for one lane.
// `p` points at (row-of-this-lane, k0) with K contiguous in memory.
// Layout (CDNA5 ISA 7.12.2, 16-bit A 16x32): VGPR0-3 hold K = half*8 .. +7,
// VGPR4-7 hold K = 16 + half*8 .. +7  (half = lane/16).  Two b128 loads.
__device__ __forceinline__ v16bf load_frag(const bf16* __restrict__ p, int half) {
    v16bf f;
    ((v8bf*)&f)[0] = *(const v8bf*)(p + half * 8);
    ((v8bf*)&f)[1] = *(const v8bf*)(p + 16 + half * 8);
    return f;
}

// Row-wise reductions over the 16 lanes of a half-wave (wave32 -> two halves).
__device__ __forceinline__ float rmax16(float x) {
#pragma unroll
    for (int m = 1; m < 16; m <<= 1) x = fmaxf(x, __shfl_xor(x, m, 16));
    return x;
}
__device__ __forceinline__ float rsum16(float x) {
#pragma unroll
    for (int m = 1; m < 16; m <<= 1) x += __shfl_xor(x, m, 16);
    return x;
}

__device__ __forceinline__ unsigned short bf16_bits(float x) {
    union { bf16 b; unsigned short u; } c;
    c.b = (bf16)x;
    return c.u;
}

// ---------------------------------------------------------------------------
// Kernel 1: f32 -> bf16 conversion (vectorized x4, grid-stride)
// ---------------------------------------------------------------------------
__global__ void cvt_f32_bf16(const float* __restrict__ src, bf16* __restrict__ dst, int n) {
    int i = (blockIdx.x * blockDim.x + threadIdx.x) * 4;
    int stride = gridDim.x * blockDim.x * 4;
    for (; i < n; i += stride) {
        float4 v = *(const float4*)(src + i);
        v4bf o;
        o[0] = (bf16)v.x; o[1] = (bf16)v.y; o[2] = (bf16)v.z; o[3] = (bf16)v.w;
        *(v4bf*)(dst + i) = o;
    }
}

// ---------------------------------------------------------------------------
// Kernel 2/5: WMMA bf16 GEMM:  C[M,N] = A[M,K] * B[N,K]^T   (K contiguous both)
// One wave computes a 32x64 output tile: 2 A fragments x 4 B fragments ->
// 8 WMMAs per k-step, ~21 FLOP/byte of fragment traffic (L2-resident weights).
// global_prefetch pulls the k+64 tiles toward the WGP during the WMMAs.
// ---------------------------------------------------------------------------
__global__ __launch_bounds__(256) void gemm_bf16(const bf16* __restrict__ A,
                                                 const bf16* __restrict__ Bm,
                                                 float* __restrict__ C,
                                                 int M, int N, int K) {
    const int lane = threadIdx.x & 31;
    const int wv   = threadIdx.x >> 5;
    const int half = lane >> 4;
    const int ln   = lane & 15;

    const int ntiles = N >> 6;                   // tiles of 64 cols
    const int tile = blockIdx.x * 8 + wv;
    const int m0 = (tile / ntiles) * 32;
    const int n0 = (tile % ntiles) * 64;

    v8f acc[2][4];
#pragma unroll
    for (int i = 0; i < 2; ++i)
#pragma unroll
        for (int j = 0; j < 4; ++j) acc[i][j] = zero8();

    const bf16* ap0 = A  + (size_t)(m0 + ln) * K;
    const bf16* ap1 = A  + (size_t)(m0 + 16 + ln) * K;
    const bf16* bp  = Bm + (size_t)(n0 + ln) * K;

    for (int k0 = 0; k0 < K; k0 += 32) {
        if (k0 + 64 < K) {                       // prefetch next-next k tiles
            __builtin_prefetch(ap0 + k0 + 64, 0, 3);
            __builtin_prefetch(ap1 + k0 + 64, 0, 3);
#pragma unroll
            for (int j = 0; j < 4; ++j)
                __builtin_prefetch(bp + (size_t)j * 16 * K + k0 + 64, 0, 3);
        }
        v16bf fa0 = load_frag(ap0 + k0, half);
        v16bf fa1 = load_frag(ap1 + k0, half);
#pragma unroll
        for (int j = 0; j < 4; ++j) {
            v16bf fb = load_frag(bp + (size_t)j * 16 * K + k0, half);
            acc[0][j] = __builtin_amdgcn_wmma_f32_16x16x32_bf16(
                false, fa0, false, fb, (short)0, acc[0][j], false, false);
            acc[1][j] = __builtin_amdgcn_wmma_f32_16x16x32_bf16(
                false, fa1, false, fb, (short)0, acc[1][j], false, false);
        }
    }

    // C layout (32-bit C/D 16x16): VGPR r holds row r + 8*half, col = ln.
#pragma unroll
    for (int i = 0; i < 2; ++i)
#pragma unroll
        for (int j = 0; j < 4; ++j)
#pragma unroll
            for (int r = 0; r < 8; ++r)
                C[(size_t)(m0 + i * 16 + r + 8 * half) * N + n0 + j * 16 + ln] =
                    acc[i][j][r];
}

// ---------------------------------------------------------------------------
// Kernel 3: per-(token, head) RMSNorm + RoPE + layout change to bf16.
// block = 128 threads = one head of one token.
//   q heads: norm * q_ln_w, rope(first 64), scale by 1/sqrt(D) -> Q[b,h,s,d]
//   k heads: norm * k_ln_w, rope(first 64)                      -> K[b,hk,s,d]
//   v heads: raw                                                -> VT[b,hk,d,s]
// ---------------------------------------------------------------------------
__global__ __launch_bounds__(128) void qkv_post(const float* __restrict__ qkv,
                                                const float* __restrict__ cosT,
                                                const float* __restrict__ sinT,
                                                const float* __restrict__ qw,
                                                const float* __restrict__ kw,
                                                bf16* __restrict__ Q,
                                                bf16* __restrict__ Kb,
                                                bf16* __restrict__ VT) {
    __shared__ float red[4];
    const int t   = threadIdx.x;
    const int hh  = blockIdx.x % NQKV;
    const int tok = blockIdx.x / NQKV;
    const int b   = tok / S_;
    const int s   = tok % S_;

    const float* row = qkv + (size_t)tok * OQKV + hh * D_;
    float v = row[t];

    float ss = v * v;
#pragma unroll
    for (int m = 1; m < 32; m <<= 1) ss += __shfl_xor(ss, m, 32);
    if ((t & 31) == 0) red[t >> 5] = ss;
    __syncthreads();
    float tot = red[0] + red[1] + red[2] + red[3];
    float inv = rsqrtf(tot * (1.0f / (float)D_) + 1e-6f);

    if (hh < H_ + HK_) {                       // q or k head: norm + rope
        const float* w = (hh < H_) ? qw : kw;
        float xn = v * inv * w[t];
        float res = xn;
        if (t < R_) {
            int   p  = (t < R_ / 2) ? t + R_ / 2 : t - R_ / 2;
            float xp = row[p] * inv * w[p];
            float c  = cosT[((size_t)b * S_ + s) * R_ + t];
            float sn = sinT[((size_t)b * S_ + s) * R_ + t];
            res = xn * c + ((t < R_ / 2) ? -xp : xp) * sn;
        }
        if (hh < H_) {
            res *= INV_SQRT_D;                 // fold score scaling into Q
            Q[(((size_t)b * H_ + hh) * S_ + s) * D_ + t] = (bf16)res;
        } else {
            Kb[(((size_t)b * HK_ + (hh - H_)) * S_ + s) * D_ + t] = (bf16)res;
        }
    } else {                                   // v head: transpose store
        VT[(((size_t)b * HK_ + (hh - H_ - HK_)) * D_ + t) * S_ + s] = (bf16)v;
    }
}

// ---------------------------------------------------------------------------
// Kernel 4: flash attention. One wave = one 16-row Q tile of one (b,h).
// Full (unmasked) softmax over all S_ keys, online max/sum.
// QK^T: A=Q(16x128), B=K-rows; PV: A=P(16x32 via LDS), B=VT rows.
// ---------------------------------------------------------------------------
__global__ __launch_bounds__(128) void flash_attn(const bf16* __restrict__ Q,
                                                  const bf16* __restrict__ Kb,
                                                  const bf16* __restrict__ VT,
                                                  bf16* __restrict__ O) {
    __shared__ unsigned short pbuf[4][16][32] __attribute__((aligned(64)));

    const int lane = threadIdx.x & 31;
    const int wv   = threadIdx.x >> 5;
    const int half = lane >> 4;
    const int ln   = lane & 15;

    const int tile = blockIdx.x * 4 + wv;        // 0 .. B*H*(S/16)-1
    const int qt = tile & (S_ / 16 - 1);
    const int h  = (tile >> 7) & (H_ - 1);
    const int b  = tile >> 11;
    const int hk = h >> 2;                        // GQA: 4 q heads per kv head
    const int q0 = qt * 16;

    // Q fragments for all of D=128 (4 chunks of K=32), kept resident.
    const bf16* qbase = Q + (((size_t)b * H_ + h) * S_ + q0 + ln) * D_;
    v16bf aq[4];
#pragma unroll
    for (int i = 0; i < 4; ++i) aq[i] = load_frag(qbase + i * 32, half);

    v8f o[8];
    float mrow[8], lrow[8];
#pragma unroll
    for (int j = 0; j < 8; ++j) o[j] = zero8();
#pragma unroll
    for (int r = 0; r < 8; ++r) { mrow[r] = -__builtin_inff(); lrow[r] = 0.0f; }

    const bf16* kbase  = Kb + ((size_t)b * HK_ + hk) * S_ * D_;
    const bf16* vtbase = VT + (((size_t)b * HK_ + hk) * D_ + ln) * S_;

    for (int kc = 0; kc < S_; kc += 32) {
        // ---- scores: two 16x16 tiles over 32 keys, K-dim = 128 ----
        v8f s0 = zero8(), s1 = zero8();
        const bf16* k0p = kbase + (size_t)(kc + ln) * D_;
        const bf16* k1p = k0p + (size_t)16 * D_;
#pragma unroll
        for (int i = 0; i < 4; ++i) {
            v16bf bk0 = load_frag(k0p + i * 32, half);
            s0 = __builtin_amdgcn_wmma_f32_16x16x32_bf16(
                false, aq[i], false, bk0, (short)0, s0, false, false);
            v16bf bk1 = load_frag(k1p + i * 32, half);
            s1 = __builtin_amdgcn_wmma_f32_16x16x32_bf16(
                false, aq[i], false, bk1, (short)0, s1, false, false);
        }

        // ---- online softmax (row stats across 16 lanes of the half) ----
        float fac[8];
#pragma unroll
        for (int r = 0; r < 8; ++r) {
            float mx = rmax16(fmaxf(s0[r], s1[r]));
            float mn = fmaxf(mrow[r], mx);
            float f  = __expf(mrow[r] - mn);
            float e0 = __expf(s0[r] - mn);
            float e1 = __expf(s1[r] - mn);
            float ps = rsum16(e0 + e1);
            lrow[r] = lrow[r] * f + ps;
            mrow[r] = mn;
            fac[r]  = f;
            const int m = r + 8 * half;          // C-layout row owned by lane
            pbuf[wv][m][ln]      = bf16_bits(e0);
            pbuf[wv][m][16 + ln] = bf16_bits(e1);
        }
#pragma unroll
        for (int j = 0; j < 8; ++j)
#pragma unroll
            for (int r = 0; r < 8; ++r) o[j][r] *= fac[r];

        // ---- reload P as an A fragment (16x32) from LDS ----
        v16bf pa = load_frag((const bf16*)&pbuf[wv][ln][0], half);

        // ---- PV: 8 output d-tiles of 16; VT gives K-contiguous B rows ----
        const bf16* vtp = vtbase + kc;
#pragma unroll
        for (int j = 0; j < 8; ++j) {
            v16bf bv = load_frag(vtp + (size_t)j * 16 * S_, half);
            o[j] = __builtin_amdgcn_wmma_f32_16x16x32_bf16(
                false, pa, false, bv, (short)0, o[j], false, false);
        }
    }

    // ---- epilogue: divide by l, store bf16 into attn[b, s, h*128 + d] ----
#pragma unroll
    for (int r = 0; r < 8; ++r) {
        float invl = 1.0f / lrow[r];
        int   row  = q0 + r + 8 * half;
        bf16* op = O + ((size_t)b * S_ + row) * (H_ * D_) + h * D_ + ln;
#pragma unroll
        for (int j = 0; j < 8; ++j) op[j * 16] = (bf16)(o[j][r] * invl);
    }
}

// ---------------------------------------------------------------------------
// Host launcher
// ---------------------------------------------------------------------------
extern "C" void kernel_launch(void* const* d_in, const int* in_sizes, int n_in,
                              void* d_out, int out_size, void* d_ws, size_t ws_size,
                              hipStream_t stream) {
    const float* hid  = (const float*)d_in[0];   // [B,S,HID]
    const float* cosT = (const float*)d_in[1];   // [B,S,64]
    const float* sinT = (const float*)d_in[2];   // [B,S,64]
    const float* wqkv = (const float*)d_in[3];   // [3072,2048]
    const float* qlw  = (const float*)d_in[4];   // [128]
    const float* klw  = (const float*)d_in[5];   // [128]
    const float* wd   = (const float*)d_in[6];   // [2048,2048]
    float* out = (float*)d_out;                  // [B,S,2048]

    char* ws = (char*)d_ws;
    bf16*  hbf    = (bf16*)(ws);                          // 16,777,216 B
    bf16*  wqkvbf = (bf16*)(ws + 16777216);               // 12,582,912 B
    bf16*  wdbf   = (bf16*)(ws + 29360128);               //  8,388,608 B
    float* qkvf   = (float*)(ws + 37748736);              // 50,331,648 B
    bf16*  qbf    = (bf16*)(ws + 88080384);               // 16,777,216 B
    bf16*  kbf    = (bf16*)(ws + 104857600);              //  4,194,304 B
    bf16*  vtbf   = (bf16*)(ws + 109051904);              //  4,194,304 B
    bf16*  attnbf = (bf16*)(ws + 113246208);              // 16,777,216 B

    // 1) convert activations / weights to bf16
    cvt_f32_bf16<<<1024, 256, 0, stream>>>(hid,  hbf,    MROWS * HID_);
    cvt_f32_bf16<<<1024, 256, 0, stream>>>(wqkv, wqkvbf, OQKV * HID_);
    cvt_f32_bf16<<<1024, 256, 0, stream>>>(wd,   wdbf,   HID_ * H_ * D_);

    // 2) QKV projection: [4096,3072] = hbf[4096,2048] * wqkv[3072,2048]^T
    {
        int tiles = (MROWS / 32) * (OQKV / 64);
        gemm_bf16<<<tiles / 8, 256, 0, stream>>>(hbf, wqkvbf, qkvf,
                                                 MROWS, OQKV, HID_);
    }

    // 3) RMSNorm + RoPE + relayout
    qkv_post<<<MROWS * NQKV, 128, 0, stream>>>(qkvf, cosT, sinT, qlw, klw,
                                               qbf, kbf, vtbf);

    // 4) flash attention (one wave per 16-row q tile)
    flash_attn<<<(B_ * H_ * (S_ / 16)) / 4, 128, 0, stream>>>(qbf, kbf, vtbf,
                                                              attnbf);

    // 5) dense projection: out[4096,2048] = attn[4096,2048] * wd[2048,2048]^T
    {
        int tiles = (MROWS / 32) * ((H_ * D_) / 64);
        gemm_bf16<<<tiles / 8, 256, 0, stream>>>(attnbf, wdbf, out,
                                                 MROWS, H_ * D_, HID_);
    }
}
